// Bidir_Attention_85512798863386
// MI455X (gfx1250) — matrix operand order repaired and checked by reading, not compiled, not run on hardware
//
#include <hip/hip_runtime.h>

// ---------------------------------------------------------------------------
// Bidirectional cross-attention with top-16 sparsified softmax, MI455X/gfx1250.
// All GEMMs run on v_wmma_f32_16x16x32_bf16 (fp32 accumulate), two independent
// accumulator chains per wave to fill WMMA hazard slots and reuse A-fragments.
// ---------------------------------------------------------------------------

#define NB   8
#define NTOK 2048
#define DIM  1024
#define TOPK 16
#define SPITCH 2052   // 2048 + 4 floats: conflict-free column access, keeps 16B row alignment

typedef __attribute__((ext_vector_type(16))) __bf16 v16bf;
typedef __attribute__((ext_vector_type(8)))  __bf16 v8bf;
typedef __attribute__((ext_vector_type(8)))  float  v8f;

static __device__ __forceinline__ unsigned short f2bf(float f) {
    __bf16 h = (__bf16)f;                 // RNE convert
    return __builtin_bit_cast(unsigned short, h);
}

static __device__ __forceinline__ v8f wmma_bf16(v16bf a, v16bf b, v8f c) {
    return __builtin_amdgcn_wmma_f32_16x16x32_bf16(false, a, false, b, (short)0, c, false, false);
}

static __device__ __forceinline__ v16bf mk16(v8bf lo, v8bf hi) {
    v16bf r;
#pragma unroll
    for (int i = 0; i < 8; ++i) { r[i] = lo[i]; r[i + 8] = hi[i]; }
    return r;
}

static __device__ __forceinline__ v16bf mk16f(float4 x0, float4 x1, float4 x2, float4 x3) {
    v16bf r;
    r[0]=(__bf16)x0.x; r[1]=(__bf16)x0.y; r[2]=(__bf16)x0.z; r[3]=(__bf16)x0.w;
    r[4]=(__bf16)x1.x; r[5]=(__bf16)x1.y; r[6]=(__bf16)x1.z; r[7]=(__bf16)x1.w;
    r[8]=(__bf16)x2.x; r[9]=(__bf16)x2.y; r[10]=(__bf16)x2.z; r[11]=(__bf16)x2.w;
    r[12]=(__bf16)x3.x; r[13]=(__bf16)x3.y; r[14]=(__bf16)x3.z; r[15]=(__bf16)x3.w;
    return r;
}

// ---------------------------------------------------------------------------
// Kernel 1: W_qkv (1024 x 3072 fp32) -> Wt (3072 x 1024 bf16, transposed)
// ---------------------------------------------------------------------------
__global__ __launch_bounds__(256) void wt_kernel(const float* __restrict__ W,
                                                 unsigned short* __restrict__ Wt) {
    int id = blockIdx.x * 256 + threadIdx.x;   // 3072*1024 threads
    int j = id >> 10;                           // output row  (0..3071)
    int k = id & 1023;                          // output col  (0..1023)
    Wt[id] = f2bf(W[k * 3072 + j]);
}

// ---------------------------------------------------------------------------
// Kernel 2: QKV projection.  C = X @ W, one wave per 16x32 strip, K=1024.
// A-fragment (fp32 loads + cvt) amortized over two independent WMMA chains.
// Q,K stored bf16 row-major (tok x d); V stored transposed (d x tok).
// ---------------------------------------------------------------------------
__global__ __launch_bounds__(256) void qkv_kernel(
    const float* __restrict__ X1, const float* __restrict__ X2,
    const unsigned short* __restrict__ Wt,
    unsigned short* __restrict__ Q1, unsigned short* __restrict__ K1, unsigned short* __restrict__ V1t,
    unsigned short* __restrict__ Q2, unsigned short* __restrict__ K2, unsigned short* __restrict__ V2t)
{
    const int lane = threadIdx.x & 31;
    const int wave = threadIdx.x >> 5;
    const int hl = lane >> 4;        // half-wave select
    const int ln = lane & 15;

    int id = blockIdx.x * 8 + wave;  // strip id < 2*8*128*96
    const int jt = id % 96;   id /= 96;    // 32-col strip (over 3072)
    const int rt = id % 128;  id /= 128;   // row tile (over 2048 tokens)
    const int b  = id % 8;    id /= 8;
    const int feat = id;                   // 0 or 1

    const float* X = (feat == 0 ? X1 : X2) + (size_t)b * NTOK * DIM;
    const int tok0 = rt * 16;
    const int j0   = jt * 32;              // 32-aligned: never straddles the 1024 q/k/v split

    const float*          arow  = X  + (size_t)(tok0 + ln) * DIM;        // A: token row
    const unsigned short* brow0 = Wt + (size_t)(j0      + ln) * DIM;     // B tile 0
    const unsigned short* brow1 = Wt + (size_t)(j0 + 16 + ln) * DIM;     // B tile 1

    v8f acc0 = {}, acc1 = {};
#pragma unroll 4
    for (int ks = 0; ks < DIM / 32; ++ks) {
        const int kbA = ks * 32 + 16 * hl;
        float4 x0 = *(const float4*)(arow + kbA);
        float4 x1 = *(const float4*)(arow + kbA + 4);
        float4 x2 = *(const float4*)(arow + kbA + 8);
        float4 x3 = *(const float4*)(arow + kbA + 12);
        v16bf A = mk16f(x0, x1, x2, x3);

        const int kbB = ks * 32 + 16 * hl;
        v8bf b00 = *(const v8bf*)(brow0 + kbB);
        v8bf b01 = *(const v8bf*)(brow0 + kbB + 8);
        v8bf b10 = *(const v8bf*)(brow1 + kbB);
        v8bf b11 = *(const v8bf*)(brow1 + kbB + 8);

        acc0 = wmma_bf16(A, mk16(b00, b01), acc0);
        acc1 = wmma_bf16(A, mk16(b10, b11), acc1);
    }

    const int which = j0 >> 10;     // 0=q 1=k 2=v (uniform across wave, same for both tiles)
    const int d0    = j0 & 1023;
    unsigned short* Qo = (feat == 0) ? Q1 : Q2;
    unsigned short* Ko = (feat == 0) ? K1 : K2;
    unsigned short* Vo = (feat == 0) ? V1t : V2t;

#pragma unroll
    for (int r = 0; r < 8; ++r) {
        const int tok = tok0 + r + 8 * hl;
        const unsigned short v0 = f2bf(acc0[r]);
        const unsigned short v1 = f2bf(acc1[r]);
        const int da = d0 + ln;
        const int db = d0 + 16 + ln;
        if (which == 0) {
            Qo[((size_t)b * NTOK + tok) * DIM + da] = v0;
            Qo[((size_t)b * NTOK + tok) * DIM + db] = v1;
        } else if (which == 1) {
            Ko[((size_t)b * NTOK + tok) * DIM + da] = v0;
            Ko[((size_t)b * NTOK + tok) * DIM + db] = v1;
        } else {
            Vo[((size_t)b * DIM + da) * NTOK + tok] = v0;
            Vo[((size_t)b * DIM + db) * NTOK + tok] = v1;
        }
    }
}

// ---------------------------------------------------------------------------
// Kernel 3: attention.  One block = (dir, batch, 16-query tile).
// S(16x2048) in LDS -> softmax stats + exact top-16 threshold -> P in place
// -> O = P @ V via WMMA.  denom == 1 in the reference (sum of softmax), so
// the output is just the top-16 probabilities times V.
// ---------------------------------------------------------------------------
__global__ __launch_bounds__(256) void attn_kernel(
    const unsigned short* __restrict__ Q1, const unsigned short* __restrict__ K1,
    const unsigned short* __restrict__ V1t,
    const unsigned short* __restrict__ Q2, const unsigned short* __restrict__ K2,
    const unsigned short* __restrict__ V2t,
    float* __restrict__ out)
{
    __shared__ float S[16][SPITCH];     // ~128 KB: scores then probabilities
    __shared__ float cand[16][256];     // per-row top-k candidates
    __shared__ float red[16][16];       // per-row reductions
    __shared__ float rmax[16], rthr[16], rinv[16];

    const int blk = blockIdx.x;         // < 2*8*128
    const int dir = blk >> 10;
    const int b   = (blk >> 7) & 7;
    const int rt  = blk & 127;
    const int tok0 = rt * 16;

    const unsigned short* Q = (dir == 0 ? Q1  : Q2 ) + (size_t)b * NTOK * DIM;
    const unsigned short* K = (dir == 0 ? K2  : K1 ) + (size_t)b * NTOK * DIM;
    const unsigned short* V = (dir == 0 ? V2t : V1t) + (size_t)b * DIM * NTOK;
    float* O = out + (size_t)dir * NB * NTOK * DIM + (size_t)b * NTOK * DIM
                   + (size_t)tok0 * DIM;

    const int lane = threadIdx.x & 31;
    const int wave = threadIdx.x >> 5;
    const int hl = lane >> 4;
    const int ln = lane & 15;

    // ---- Phase 1: S = (Q K^T) * 1/sqrt(D), two key tiles per A-fragment ----
    const unsigned short* qrow = Q + (size_t)(tok0 + ln) * DIM;
    for (int tp = wave; tp < NTOK / 32; tp += 8) {
        const int t0 = tp * 2;
        const unsigned short* krow0 = K + (size_t)(t0 * 16      + ln) * DIM;
        const unsigned short* krow1 = K + (size_t)(t0 * 16 + 16 + ln) * DIM;
        v8f acc0 = {}, acc1 = {};
#pragma unroll 4
        for (int ks = 0; ks < DIM / 32; ++ks) {
            const int kbA = ks * 32 + 16 * hl;
            v8bf a0 = *(const v8bf*)(qrow + kbA);
            v8bf a1 = *(const v8bf*)(qrow + kbA + 8);
            v16bf A = mk16(a0, a1);
            const int kbB = ks * 32 + 16 * hl;
            v8bf b00 = *(const v8bf*)(krow0 + kbB);
            v8bf b01 = *(const v8bf*)(krow0 + kbB + 8);
            v8bf b10 = *(const v8bf*)(krow1 + kbB);
            v8bf b11 = *(const v8bf*)(krow1 + kbB + 8);
            acc0 = wmma_bf16(A, mk16(b00, b01), acc0);
            acc1 = wmma_bf16(A, mk16(b10, b11), acc1);
        }
#pragma unroll
        for (int r = 0; r < 8; ++r) {
            S[r + 8 * hl][t0 * 16      + ln] = acc0[r] * 0.03125f;
            S[r + 8 * hl][t0 * 16 + 16 + ln] = acc1[r] * 0.03125f;
        }
    }
    __syncthreads();

    // ---- Phase 2: per-row max, top-16 threshold, sum(exp) ----
    const int ty = threadIdx.x >> 4;    // row 0..15
    const int tx = threadIdx.x & 15;    // 16 threads per row

    float l[TOPK];
#pragma unroll
    for (int i = 0; i < TOPK; ++i) l[i] = -3.4e38f;
    for (int j = 0; j < NTOK / 16; ++j) {
        float v = S[ty][tx + 16 * j];
        if (v > l[0]) {
            l[0] = v;
#pragma unroll
            for (int i = 0; i < TOPK - 1; ++i)
                if (l[i] > l[i + 1]) { float t = l[i]; l[i] = l[i + 1]; l[i + 1] = t; }
        }
    }
#pragma unroll
    for (int i = 0; i < TOPK; ++i) cand[ty][tx * TOPK + i] = l[i];
    __syncthreads();

    if (tx == 0) {
        float g[TOPK];
#pragma unroll
        for (int i = 0; i < TOPK; ++i) g[i] = -3.4e38f;
        for (int j = 0; j < 256; ++j) {
            float v = cand[ty][j];
            if (v > g[0]) {
                g[0] = v;
#pragma unroll
                for (int i = 0; i < TOPK - 1; ++i)
                    if (g[i] > g[i + 1]) { float t = g[i]; g[i] = g[i + 1]; g[i + 1] = t; }
            }
        }
        rthr[ty] = g[0];            // 16th largest score
        rmax[ty] = g[TOPK - 1];     // row max
    }
    __syncthreads();

    const float M = rmax[ty];
    float s = 0.f;
    for (int j = 0; j < NTOK / 16; ++j)
        s += __expf(S[ty][tx + 16 * j] - M);
    red[ty][tx] = s;
    __syncthreads();
    if (tx == 0) {
        float z = 0.f;
        for (int j = 0; j < 16; ++j) z += red[ty][j];
        rinv[ty] = 1.f / z;
    }
    __syncthreads();

    // ---- Phase 2.5: P = mask(top16) * exp(S - M) / Z, in place ----
    const float T = rthr[ty];
    const float iz = rinv[ty];
    for (int j = 0; j < NTOK / 16; ++j) {
        const int c = tx + 16 * j;
        float v = S[ty][c];
        S[ty][c] = (v >= T) ? __expf(v - M) * iz : 0.f;
    }
    __syncthreads();

    // ---- Phase 3: O = P @ V, two d-tiles per P-fragment (V transposed d x tok) ----
    for (int dp = wave; dp < DIM / 32; dp += 8) {
        const int d0 = dp * 32;
        const unsigned short* vrow0 = V + (size_t)(d0      + ln) * NTOK;
        const unsigned short* vrow1 = V + (size_t)(d0 + 16 + ln) * NTOK;
        v8f acc0 = {}, acc1 = {};
#pragma unroll 2
        for (int ks = 0; ks < NTOK / 32; ++ks) {
            const int kbA = ks * 32 + 16 * hl;
            const float4* p0 = (const float4*)&S[ln][kbA];
            float4 x0 = p0[0], x1 = p0[1];
            const float4* p1 = (const float4*)&S[ln][kbA + 8];
            float4 x2 = p1[0], x3 = p1[1];
            v16bf A = mk16f(x0, x1, x2, x3);
            const int kbB = ks * 32 + 16 * hl;
            v8bf b00 = *(const v8bf*)(vrow0 + kbB);
            v8bf b01 = *(const v8bf*)(vrow0 + kbB + 8);
            v8bf b10 = *(const v8bf*)(vrow1 + kbB);
            v8bf b11 = *(const v8bf*)(vrow1 + kbB + 8);
            acc0 = wmma_bf16(A, mk16(b00, b01), acc0);
            acc1 = wmma_bf16(A, mk16(b10, b11), acc1);
        }
#pragma unroll
        for (int r = 0; r < 8; ++r) {
            O[(size_t)(r + 8 * hl) * DIM + d0      + ln] = acc0[r];
            O[(size_t)(r + 8 * hl) * DIM + d0 + 16 + ln] = acc1[r];
        }
    }
}

// ---------------------------------------------------------------------------
extern "C" void kernel_launch(void* const* d_in, const int* in_sizes, int n_in,
                              void* d_out, int out_size, void* d_ws, size_t ws_size,
                              hipStream_t stream)
{
    (void)in_sizes; (void)n_in; (void)out_size; (void)ws_size;

    const float* f1 = (const float*)d_in[0];
    const float* f2 = (const float*)d_in[1];
    const float* W  = (const float*)d_in[2];
    float* out = (float*)d_out;

    char* ws = (char*)d_ws;
    const size_t QKV_BYTES = (size_t)NB * NTOK * DIM * sizeof(unsigned short); // 32 MiB
    unsigned short* Wt  = (unsigned short*)ws; ws += (size_t)3 * DIM * DIM * sizeof(unsigned short);
    unsigned short* Q1  = (unsigned short*)ws; ws += QKV_BYTES;
    unsigned short* K1  = (unsigned short*)ws; ws += QKV_BYTES;
    unsigned short* V1t = (unsigned short*)ws; ws += QKV_BYTES;
    unsigned short* Q2  = (unsigned short*)ws; ws += QKV_BYTES;
    unsigned short* K2  = (unsigned short*)ws; ws += QKV_BYTES;
    unsigned short* V2t = (unsigned short*)ws; ws += QKV_BYTES;

    wt_kernel<<<(3 * DIM * DIM) / 256, 256, 0, stream>>>(W, Wt);

    // 2 feats * 8 batches * 128 row tiles * 96 col strips (32 wide), 8 strips per block
    qkv_kernel<<<(2 * 8 * 128 * 96) / 8, 256, 0, stream>>>(
        f1, f2, Wt, Q1, K1, V1t, Q2, K2, V2t);

    attn_kernel<<<2 * NB * (NTOK / 16), 256, 0, stream>>>(
        Q1, K1, V1t, Q2, K2, V2t, out);
}
